// MultiHeadLatentAttention_11785390260753
// MI455X (gfx1250) — matrix operand order repaired and checked by reading, not compiled, not run on hardware
//
#include <hip/hip_runtime.h>
#include <hip/hip_bf16.h>

// ---------------------------------------------------------------------------
// MLA forward for gfx1250 (CDNA5): bf16 WMMA + async global->LDS staging.
// ---------------------------------------------------------------------------

typedef __attribute__((ext_vector_type(16))) __bf16 v16bf;
typedef __attribute__((ext_vector_type(8)))  float  v8f;

#define B_DIM    2
#define T_DIM    2048
#define C_DIM    2048
#define H_DIM    16
#define HD_DIM   128
#define LORA_DIM 512
#define RD_DIM   64
#define ND_DIM   64
#define M_DIM    (B_DIM * T_DIM)          // 4096 tokens
#define KVN_DIM  (LORA_DIM + RD_DIM)      // 576
#define HND_DIM  (H_DIM * ND_DIM)         // 1024

static __device__ __forceinline__ unsigned short f2bf(float f) {
  unsigned int u = __float_as_uint(f);
  u += 0x7FFFu + ((u >> 16) & 1u);        // round-to-nearest-even
  return (unsigned short)(u >> 16);
}

union Frag {                               // 8 dwords <-> 16 bf16 (one WMMA A/B operand)
  unsigned int u[8];
  v16bf        v;
};

// Async DMA of 16 bytes global -> LDS, tracked by ASYNCcnt (no VGPR staging).
// VDST VGPR holds the wave-relative LDS byte offset == low 32 bits of the
// generic pointer to a __shared__ object (flat LDS aperture encoding).
static __device__ __forceinline__ void async_ld_b128(void* lds_ptr, const void* gptr) {
  unsigned lds_off = (unsigned)(size_t)lds_ptr;
  asm volatile("global_load_async_to_lds_b128 %0, %1, off"
               :: "v"(lds_off), "v"(gptr) : "memory");
}
static __device__ __forceinline__ void wait_async0() {
  asm volatile("s_wait_asynccnt 0x0" ::: "memory");
}

// ---------------------------------------------------------------------------
// Elementwise converts
// ---------------------------------------------------------------------------
__global__ void k_convert_bf16(const float* __restrict__ in,
                               unsigned short* __restrict__ out, int n) {
  int i = blockIdx.x * blockDim.x + threadIdx.x;
  if (i < n) out[i] = f2bf(in[i]);
}

// in: [K,N] f32 row-major  ->  out: [N,K] bf16 row-major (weights, done once)
__global__ void k_transpose_bf16(const float* __restrict__ in,
                                 unsigned short* __restrict__ out,
                                 int K, int N) {
  long long i = (long long)blockIdx.x * blockDim.x + threadIdx.x;
  long long total = (long long)K * N;
  if (i >= total) return;
  int n = (int)(i / K);
  int k = (int)(i % K);
  out[i] = f2bf(in[(long long)k * N + n]);
}

// first LORA columns of ckv -> bf16 latent activations
__global__ void k_extract_latent(const float* __restrict__ ckv,
                                 unsigned short* __restrict__ latent, int n) {
  int i = blockIdx.x * blockDim.x + threadIdx.x;
  if (i >= n) return;
  int row = i / LORA_DIM, col = i % LORA_DIM;
  latent[i] = f2bf(ckv[(long long)row * KVN_DIM + col]);
}

// ---------------------------------------------------------------------------
// GEMM: C[M,N] f32 = A[M,K] bf16 (row-major) @ Bt[N,K] bf16 (pre-transposed)
//
// Block = 128 threads (4 waves), 64(M) x 64(N) macro-tile, K stepped by 32.
// Per k-step both 64x32 bf16 tiles (A rows, Bt rows) are DMA'd into
// double-buffered LDS with global_load_async_to_lds_b128 (ASYNCcnt), then
// each wave feeds v_wmma_f32_16x16x32_bf16 from ds_load_b128 fragments.
//
// WMMA layouts (D per ISA 7.12.2; A/B k order chosen by this code):
//   A 16x32 bf16 : lane row=l&15, half=l>>4; VGPR v: K=half*16+2v (contiguous dwords; this code's order, not ISA table)
//   B 32x16 bf16 : lane n=l&15;  VGPR v: K=(l>>4)*16+2v  (contiguous dwords; this code's order, not ISA table)
//   D 16x16 f32  : VGPR r: row=r+(l>>4)*8, col=l&15
// ---------------------------------------------------------------------------
__global__ void __launch_bounds__(128)
k_gemm_bf16(const unsigned short* __restrict__ A,
            const unsigned short* __restrict__ Bt,
            float* __restrict__ C, int M, int N, int K) {
  __shared__ unsigned short Atile[2][64][32];   // 8 KB
  __shared__ unsigned short Btile[2][64][32];   // 8 KB

  const int tid  = threadIdx.x;
  const int wave = tid >> 5;
  const int lane = tid & 31;
  const int lrow = lane & 15;
  const int half = lane >> 4;
  const int m0 = blockIdx.y * 64;
  const int n0 = blockIdx.x * 64;

  const unsigned short* Abase = A  + (size_t)m0 * K;
  const unsigned short* Bbase = Bt + (size_t)n0 * K;

  // copy 64 rows x 32 ush (64B) per tile: 256 x 16B chunks, 2 per lane
  const int c0row = (tid * 2) >> 2, c0chk = (tid * 2) & 3;
  const int c1row = (tid * 2 + 1) >> 2, c1chk = (tid * 2 + 1) & 3;

  auto stage = [&](int k0, int buf) {
    async_ld_b128(&Atile[buf][c0row][c0chk * 8],
                  Abase + (size_t)c0row * K + k0 + c0chk * 8);
    async_ld_b128(&Atile[buf][c1row][c1chk * 8],
                  Abase + (size_t)c1row * K + k0 + c1chk * 8);
    async_ld_b128(&Btile[buf][c0row][c0chk * 8],
                  Bbase + (size_t)c0row * K + k0 + c0chk * 8);
    async_ld_b128(&Btile[buf][c1row][c1chk * 8],
                  Bbase + (size_t)c1row * K + k0 + c1chk * 8);
  };

  v8f acc0 = {}, acc1 = {}, acc2 = {}, acc3 = {};

  const int ksteps = K >> 5;
  stage(0, 0);

  for (int ks = 0; ks < ksteps; ++ks) {
    const int buf = ks & 1;
    wait_async0();          // my async writes done
    __syncthreads();        // everyone's async writes visible

    if (ks + 1 < ksteps) stage((ks + 1) << 5, buf ^ 1);

    // A fragment from LDS (two 16B chunks -> ds_load_b128)
    Frag a;
    const unsigned int* Arow = (const unsigned int*)&Atile[buf][wave * 16 + lrow][0];
#pragma unroll
    for (int v = 0; v < 8; ++v) {
      int kk = (half << 4) + (v << 1);
      a.u[v] = Arow[kk >> 1];
    }
#pragma unroll
    for (int nt = 0; nt < 4; ++nt) {
      Frag b;
      const unsigned int* Brow = (const unsigned int*)&Btile[buf][nt * 16 + lrow][0];
#pragma unroll
      for (int v = 0; v < 8; ++v) b.u[v] = Brow[half * 8 + v];
      v8f* acc = (nt == 0) ? &acc0 : (nt == 1) ? &acc1 : (nt == 2) ? &acc2 : &acc3;
      *acc = __builtin_amdgcn_wmma_f32_16x16x32_bf16(false, a.v, false, b.v,
                                                     (short)0, *acc, false, false);
    }
    __syncthreads();        // reads of buf complete before it is re-staged
  }

#pragma unroll
  for (int r = 0; r < 8; ++r) {
    size_t off = (size_t)(m0 + wave * 16 + r + half * 8) * N + n0 + lrow;
    C[off]      = acc0[r];
    C[off + 16] = acc1[r];
    C[off + 32] = acc2[r];
    C[off + 48] = acc3[r];
  }
}

// ---------------------------------------------------------------------------
// RoPE + pack:
//   qb [B*H, T, 128] bf16 : [q_nope | rope(q_rope)]
//   kb [B*H, T, 128] bf16 : [kv      | rope(k_rope broadcast over H)]
//   vbT[B*H, 128, T] bf16 : [kv      | 0]  (transposed so AV B-frags are dwords)
// ---------------------------------------------------------------------------
__global__ void k_rope_pack(const float* __restrict__ q,     // [B,T,H*HD]
                            const float* __restrict__ kv,    // [B,T,H*ND]
                            const float* __restrict__ ckv,   // [B,T,LORA+RD]
                            unsigned short* __restrict__ qb,
                            unsigned short* __restrict__ kb,
                            unsigned short* __restrict__ vbT) {
  const int d = threadIdx.x;        // 0..127
  const int t = blockIdx.x;
  const int h = blockIdx.y;
  const int b = blockIdx.z;
  const size_t tok = (size_t)b * T_DIM + t;
  const size_t bh  = (size_t)b * H_DIM + h;
  const size_t qkrow = (bh * T_DIM + t) * HD_DIM;

  float qo, ko, vo;
  if (d < ND_DIM) {
    qo = q[tok * (H_DIM * HD_DIM) + h * HD_DIM + d];
    float kvv = kv[tok * HND_DIM + h * ND_DIM + d];
    ko = kvv;
    vo = kvv;
  } else {
    int j = d - ND_DIM;
    int i = j >> 1, odd = j & 1;
    // freq_i = theta^(-2i/RD); log(1e5) = 11.512925...
    float freq = __expf(-((float)(2 * i) / (float)RD_DIM) * 11.512925464970229f);
    float ang = (float)t * freq;
    float c = cosf(ang), s = sinf(ang);

    size_t qbase = tok * (H_DIM * HD_DIM) + h * HD_DIM + ND_DIM + 2 * i;
    float qx0 = q[qbase], qx1 = q[qbase + 1];
    size_t kbase = tok * KVN_DIM + LORA_DIM + 2 * i;
    float kx0 = ckv[kbase], kx1 = ckv[kbase + 1];

    qo = odd ? (qx0 * s + qx1 * c) : (qx0 * c - qx1 * s);
    ko = odd ? (kx0 * s + kx1 * c) : (kx0 * c - kx1 * s);
    vo = 0.0f;
  }
  qb[qkrow + d] = f2bf(qo);
  kb[qkrow + d] = f2bf(ko);
  vbT[(bh * HD_DIM + d) * T_DIM + t] = f2bf(vo);
}

// ---------------------------------------------------------------------------
// Flash attention, causal. One wave per (b,h, 16-query tile).
// Key tiles of 32: S(16x32) = two 16x16 WMMAs x 4 feature steps,
// online softmax (16-lane xor butterflies), P re-layout via per-wave LDS
// (D-layout -> A-layout) with s_wait_dscnt, O += P(16x32) @ V(32x128)
// as 8 WMMAs into 8 f32 accumulators.
// ---------------------------------------------------------------------------
__global__ void __launch_bounds__(32)
k_mla_flash(const unsigned short* __restrict__ qb,
            const unsigned short* __restrict__ kb,
            const unsigned short* __restrict__ vbT,
            unsigned short* __restrict__ yb) {    // [B,T,H*HD] bf16
  __shared__ unsigned short pl[16 * 32];

  const int lane = threadIdx.x & 31;
  const int lrow = lane & 15;
  const int half = lane >> 4;
  const int t0 = blockIdx.x * 16;
  const int h = blockIdx.y, b = blockIdx.z;
  const size_t bh = (size_t)b * H_DIM + h;
  const float scale = 0.088388347648318447f;      // 1/sqrt(128)

  // Q fragments for all 4 feature steps (16 rows x 32 features each)
  unsigned int qa[4][8];
  const unsigned int* Qrow = (const unsigned int*)(qb + (bh * T_DIM + t0 + lrow) * HD_DIM);
#pragma unroll
  for (int kk = 0; kk < 4; ++kk)
#pragma unroll
    for (int v = 0; v < 8; ++v) {
      int f = kk * 32 + (half << 4) + (v << 1);
      qa[kk][v] = Qrow[f >> 1];
    }

  v8f o[8];
#pragma unroll
  for (int nt = 0; nt < 8; ++nt) o[nt] = {};
  float mrun[8], lrun[8];
#pragma unroll
  for (int r = 0; r < 8; ++r) { mrun[r] = -3.0e38f; lrun[r] = 0.0f; }

  for (int s0 = 0; s0 <= t0 + 15; s0 += 32) {
    // ---- scores: S = Q @ K^T for keys [s0, s0+32) ----
    v8f sa0 = {}, sa1 = {};
    const unsigned int* K0 = (const unsigned int*)(kb + (bh * T_DIM + s0 + lrow) * HD_DIM);
    const unsigned int* K1 = (const unsigned int*)(kb + (bh * T_DIM + s0 + 16 + lrow) * HD_DIM);
#pragma unroll
    for (int kk = 0; kk < 4; ++kk) {
      const int kh2 = (kk * 32 + half * 16) >> 1;
      Frag aq, b0, b1;
#pragma unroll
      for (int v = 0; v < 8; ++v) {
        aq.u[v] = qa[kk][v];
        b0.u[v] = K0[kh2 + v];
        b1.u[v] = K1[kh2 + v];
      }
      sa0 = __builtin_amdgcn_wmma_f32_16x16x32_bf16(false, aq.v, false, b0.v, (short)0, sa0, false, false);
      sa1 = __builtin_amdgcn_wmma_f32_16x16x32_bf16(false, aq.v, false, b1.v, (short)0, sa1, false, false);
    }

    // ---- scale, causal mask, online softmax ----
    float p0[8], p1[8], corr[8];
#pragma unroll
    for (int r = 0; r < 8; ++r) {
      int qidx = t0 + r + half * 8;
      float v0 = sa0[r] * scale;
      float v1 = sa1[r] * scale;
      if (s0 + lrow > qidx)      v0 = -3.0e38f;
      if (s0 + 16 + lrow > qidx) v1 = -3.0e38f;
      float rm = fmaxf(v0, v1);
      rm = fmaxf(rm, __shfl_xor(rm, 1));
      rm = fmaxf(rm, __shfl_xor(rm, 2));
      rm = fmaxf(rm, __shfl_xor(rm, 4));
      rm = fmaxf(rm, __shfl_xor(rm, 8));
      float mnew = fmaxf(mrun[r], rm);
      float cor = __expf(mrun[r] - mnew);
      float e0 = __expf(v0 - mnew);
      float e1 = __expf(v1 - mnew);
      float rs = e0 + e1;
      rs += __shfl_xor(rs, 1);
      rs += __shfl_xor(rs, 2);
      rs += __shfl_xor(rs, 4);
      rs += __shfl_xor(rs, 8);
      lrun[r] = lrun[r] * cor + rs;
      mrun[r] = mnew;
      corr[r] = cor;
      p0[r] = e0;
      p1[r] = e1;
    }
#pragma unroll
    for (int nt = 0; nt < 8; ++nt)
#pragma unroll
      for (int r = 0; r < 8; ++r) o[nt][r] *= corr[r];

    // ---- P: D-layout -> A-layout via per-wave LDS ----
#pragma unroll
    for (int r = 0; r < 8; ++r) {
      pl[(r + half * 8) * 32 + lrow]      = f2bf(p0[r]);
      pl[(r + half * 8) * 32 + 16 + lrow] = f2bf(p1[r]);
    }
    asm volatile("s_wait_dscnt 0x0" ::: "memory");
    Frag pa;
    const unsigned int* Prow = (const unsigned int*)pl + lrow * 16;
#pragma unroll
    for (int v = 0; v < 8; ++v) {
      int kk = (half << 4) + (v << 1);
      pa.u[v] = Prow[kk >> 1];
    }

    // ---- O += P(16x32) @ V(32x128): 8 WMMAs over feature subtiles ----
#pragma unroll
    for (int nt = 0; nt < 8; ++nt) {
      Frag vf;
      const unsigned int* Vrow =
          (const unsigned int*)(vbT + (bh * HD_DIM + nt * 16 + lrow) * T_DIM + s0);
#pragma unroll
      for (int v = 0; v < 8; ++v) vf.u[v] = Vrow[half * 8 + v];
      o[nt] = __builtin_amdgcn_wmma_f32_16x16x32_bf16(false, pa.v, false, vf.v, (short)0, o[nt], false, false);
    }
  }

  // ---- epilogue: y = O / l, stored token-major bf16 for the Wo GEMM ----
#pragma unroll
  for (int nt = 0; nt < 8; ++nt)
#pragma unroll
    for (int r = 0; r < 8; ++r) {
      float y = o[nt][r] / lrun[r];
      yb[((size_t)b * T_DIM + t0 + r + half * 8) * (H_DIM * HD_DIM)
         + h * HD_DIM + nt * 16 + lrow] = f2bf(y);
    }
}

// ---------------------------------------------------------------------------
// Host-side launch sequence (graph-capturable: no mallocs, no syncs)
// ---------------------------------------------------------------------------
extern "C" void kernel_launch(void* const* d_in, const int* in_sizes, int n_in,
                              void* d_out, int out_size, void* d_ws, size_t ws_size,
                              hipStream_t stream) {
  (void)in_sizes; (void)n_in; (void)out_size; (void)ws_size;

  const float* x    = (const float*)d_in[0];   // [B,T,C]
  const float* Wq   = (const float*)d_in[1];   // [C, H*HD]
  const float* Wkva = (const float*)d_in[2];   // [C, LORA+RD]
  const float* Wkvb = (const float*)d_in[3];   // [LORA, H*ND]
  const float* Wo   = (const float*)d_in[4];   // [H*HD, C]
  float* out = (float*)d_out;                  // [B,T,C]

  char* p = (char*)d_ws;
  auto alloc = [&](size_t bytes) -> char* {
    char* r = p;
    p += (bytes + 255) & ~(size_t)255;
    return r;
  };

  unsigned short* xb    = (unsigned short*)alloc((size_t)M_DIM * C_DIM * 2);
  unsigned short* Wqt   = (unsigned short*)alloc((size_t)C_DIM * C_DIM * 2);     // [2048,2048]
  unsigned short* Wkvat = (unsigned short*)alloc((size_t)KVN_DIM * C_DIM * 2);   // [576,2048]
  unsigned short* Wkvbt = (unsigned short*)alloc((size_t)HND_DIM * LORA_DIM * 2);// [1024,512]
  unsigned short* Wot   = (unsigned short*)alloc((size_t)C_DIM * C_DIM * 2);     // [2048,2048]
  float*          qf    = (float*)alloc((size_t)M_DIM * C_DIM * 4);
  float*          ckv   = (float*)alloc((size_t)M_DIM * KVN_DIM * 4);
  unsigned short* latb  = (unsigned short*)alloc((size_t)M_DIM * LORA_DIM * 2);
  float*          kvf   = (float*)alloc((size_t)M_DIM * HND_DIM * 4);
  unsigned short* qbb   = (unsigned short*)alloc((size_t)M_DIM * C_DIM * 2);     // [BH,T,128]
  unsigned short* kbb   = (unsigned short*)alloc((size_t)M_DIM * C_DIM * 2);     // [BH,T,128]
  unsigned short* vbt   = (unsigned short*)alloc((size_t)M_DIM * C_DIM * 2);     // [BH,128,T]
  unsigned short* ybb   = (unsigned short*)alloc((size_t)M_DIM * C_DIM * 2);     // [B,T,H*HD]

  // 1) converts (weights transposed so B-frags are contiguous K-pairs)
  {
    int n = M_DIM * C_DIM;
    k_convert_bf16<<<(n + 255) / 256, 256, 0, stream>>>(x, xb, n);
  }
  {
    long long tot = (long long)C_DIM * C_DIM;
    k_transpose_bf16<<<(unsigned)((tot + 255) / 256), 256, 0, stream>>>(Wq, Wqt, C_DIM, C_DIM);
  }
  {
    long long tot = (long long)C_DIM * KVN_DIM;
    k_transpose_bf16<<<(unsigned)((tot + 255) / 256), 256, 0, stream>>>(Wkva, Wkvat, C_DIM, KVN_DIM);
  }
  {
    long long tot = (long long)LORA_DIM * HND_DIM;
    k_transpose_bf16<<<(unsigned)((tot + 255) / 256), 256, 0, stream>>>(Wkvb, Wkvbt, LORA_DIM, HND_DIM);
  }
  {
    long long tot = (long long)C_DIM * C_DIM;
    k_transpose_bf16<<<(unsigned)((tot + 255) / 256), 256, 0, stream>>>(Wo, Wot, C_DIM, C_DIM);
  }

  // 2) q = x @ Wq               (4096 x 2048 x 2048)
  k_gemm_bf16<<<dim3(C_DIM / 64, M_DIM / 64), 128, 0, stream>>>(xb, Wqt, qf, M_DIM, C_DIM, C_DIM);
  // 3) ckv = x @ Wkva           (4096 x 576 x 2048)
  k_gemm_bf16<<<dim3(KVN_DIM / 64, M_DIM / 64), 128, 0, stream>>>(xb, Wkvat, ckv, M_DIM, KVN_DIM, C_DIM);
  // 4) latent -> bf16
  {
    int n = M_DIM * LORA_DIM;
    k_extract_latent<<<(n + 255) / 256, 256, 0, stream>>>(ckv, latb, n);
  }
  // 5) kv = latent @ Wkvb       (4096 x 1024 x 512)
  k_gemm_bf16<<<dim3(HND_DIM / 64, M_DIM / 64), 128, 0, stream>>>(latb, Wkvbt, kvf, M_DIM, HND_DIM, LORA_DIM);
  // 6) RoPE + pack q/k/v
  k_rope_pack<<<dim3(T_DIM, H_DIM, B_DIM), HD_DIM, 0, stream>>>(qf, kvf, ckv, qbb, kbb, vbt);
  // 7) causal flash attention
  k_mla_flash<<<dim3(T_DIM / 16, H_DIM, B_DIM), 32, 0, stream>>>(qbb, kbb, vbt, ybb);
  // 8) out = y @ Wo             (4096 x 2048 x 2048), f32 output
  k_gemm_bf16<<<dim3(C_DIM / 64, M_DIM / 64), 128, 0, stream>>>(ybb, Wot, out, M_DIM, C_DIM, C_DIM);
}
